// FixedEGNNLayer_18580028522962
// MI455X (gfx1250) — matrix operand order repaired and checked by reading, not compiled, not run on hardware
//
#include <hip/hip_runtime.h>
#include <hip/hip_bf16.h>

typedef _Float16 f16;
typedef _Float16 v16h __attribute__((ext_vector_type(16)));
typedef _Float16 v8h  __attribute__((ext_vector_type(8)));
typedef float    v8f  __attribute__((ext_vector_type(8)));

#define WMMA_F16(A,B,C) __builtin_amdgcn_wmma_f32_16x16x32_f16(false,(A),false,(B),(short)0,(C),false,false)

__device__ __forceinline__ float silu_f(float v){ return v / (1.0f + __expf(-v)); }

__device__ __forceinline__ void atomic_add_f32(float* p, float v){
  __hip_atomic_fetch_add(p, v, __ATOMIC_RELAXED, __HIP_MEMORY_SCOPE_AGENT);
}

// A fragment (16x32 f16) for one 16-row tile from LDS row-major [16][ldA]
// lanes 0-15: row M=lane, elems 0..7 -> K=k0+0..7, elems 8..15 -> K=k0+16..23
// lanes 16-31: row M=lane-16, elems 0..7 -> K=k0+8..15, elems 8..15 -> K=k0+24..31
__device__ __forceinline__ v16h load_a_frag(const f16* row, int kc, int lane){
  const int half = (lane >> 4) & 1;
  const int k0 = kc << 5;
  v8h lo = *(const v8h*)(row + k0 + half * 8);
  v8h hi = *(const v8h*)(row + k0 + 16 + half * 8);
  v16h r;
#pragma unroll
  for (int i = 0; i < 8; ++i) { r[i] = lo[i]; r[i + 8] = hi[i]; }
  return r;
}

// B fragment (32x16 f16): pre-swizzled so each lane reads 16 contiguous halves (32B)
__device__ __forceinline__ v16h load_b_frag(const f16* Bsw, int numKC, int nt, int kc, int lane){
  return *(const v16h*)(Bsw + ((((size_t)nt * numKC + kc) * 32 + lane) << 4));
}

// ---------- prep kernels ----------
__global__ void conv_h_kernel(const float* __restrict__ h, f16* __restrict__ hh, int n){
  int i = blockIdx.x * blockDim.x + threadIdx.x;
  if (i < n) hh[i] = (f16)h[i];
}

// swizzle W [Krows][128] (row-major in,out) into B-fragment order, zero-padded K
__global__ void swizzle_kernel(const float* __restrict__ W, f16* __restrict__ Wsw,
                               int Krows, int numKC){
  int idx = blockIdx.x * blockDim.x + threadIdx.x;
  if (idx >= numKC * 4096) return;            // 8 ntiles * numKC * 32 lanes * 16 elems
  int e    = idx & 15;
  int lane = (idx >> 4) & 31;
  int rem  = idx >> 9;
  int kc   = rem % numKC;
  int nt   = rem / numKC;
  int n = nt * 16 + (lane & 15);
  int k = kc * 32 + ((e >> 3) << 4) + ((lane >> 4) & 1) * 8 + (e & 7);
  float v = (k < Krows) ? W[(size_t)k * 128 + n] : 0.0f;
  Wsw[idx] = (f16)v;
}

// ---------- edge kernel: phi_e, phi_x, scatter ----------
__global__ __launch_bounds__(256)
void egnn_edge_kernel(const f16* __restrict__ hh,
                      const float* __restrict__ x,
                      const float* __restrict__ vel,
                      const long long* __restrict__ eidx,
                      const f16* __restrict__ We1sw, const float* __restrict__ be1,
                      const f16* __restrict__ We2sw, const float* __restrict__ be2,
                      const f16* __restrict__ Wx1sw, const float* __restrict__ bx1,
                      const float* __restrict__ Wx2,  const float* __restrict__ bx2,
                      float* __restrict__ agg, float* __restrict__ num,
                      float* __restrict__ cnt, float* __restrict__ out_m, int E)
{
  __shared__ __align__(16) f16 sF[32 * 288];   // features [32][288]
  __shared__ __align__(16) f16 sY[32 * 128];   // phi_e layer1 out
  __shared__ __align__(16) f16 sM[32 * 128];   // m_ij (f16)
  __shared__ __align__(16) f16 sX[32 * 128];   // phi_x layer1 out
  __shared__ float sRel[32 * 3];
  __shared__ float sCW[32];
  __shared__ int sSrc[32], sDst[32];

  const int tid  = threadIdx.x;
  const int lane = tid & 31;
  const int w    = tid >> 5;          // wave id = ntile (0..7)
  const int e0   = blockIdx.x * 32;

  // ---- geometry (one thread per edge) ----
  if (tid < 32) {
    int e = e0 + tid;
    long long s = 0, d = 0;
    if (e < E) { s = eidx[e]; d = eidx[(size_t)E + e]; }
    sSrc[tid] = (int)s; sDst[tid] = (int)d;
    float rx = x[s*3+0] - x[d*3+0];
    float ry = x[s*3+1] - x[d*3+1];
    float rz = x[s*3+2] - x[d*3+2];
    float d2 = rx*rx + ry*ry + rz*rz;
    float dist = fmaxf(sqrtf(d2), 1e-8f);
    float inv = 1.0f / dist;
    float hx = rx*inv, hy = ry*inv, hz = rz*inv;
    sRel[tid*3+0] = rx; sRel[tid*3+1] = ry; sRel[tid*3+2] = rz;
    f16* fr = sF + tid * 288;
    fr[256] = (f16)d2;
#pragma unroll
    for (int a = 0; a < 5; ++a){
      const float* vs = vel + ((size_t)s * 5 + a) * 3;
      const float* vd = vel + ((size_t)d * 5 + a) * 3;
      fr[257 + a] = (f16)(vs[0]*hx + vs[1]*hy + vs[2]*hz);
      fr[262 + a] = (f16)(vd[0]*hx + vd[1]*hy + vd[2]*hz);
    }
    for (int k = 267; k < 288; ++k) fr[k] = (f16)0.0f;
    if (e >= E) for (int k = 256; k < 267; ++k) fr[k] = (f16)0.0f;
  }
  __syncthreads();

  // ---- gather h halves: 8 threads per edge, 16 halves (32B) each ----
  {
    int edge = tid >> 3, part = tid & 7;
    const uint4* ps = (const uint4*)(hh + (size_t)sSrc[edge] * 128 + part * 16);
    const uint4* pd = (const uint4*)(hh + (size_t)sDst[edge] * 128 + part * 16);
    uint4* q0 = (uint4*)(sF + edge * 288 + part * 16);
    uint4* q1 = (uint4*)(sF + edge * 288 + 128 + part * 16);
    q0[0] = ps[0]; q0[1] = ps[1];
    q1[0] = pd[0]; q1[1] = pd[1];
  }
  __syncthreads();

  const int col = lane & 15;
  const int half = (lane >> 4) & 1;
  const int n = w * 16 + col;

  // ---- phi_e layer 1: [32,288] x [288,128] ----
  {
    v8f acc0 = {}, acc1 = {};
#pragma unroll
    for (int kc = 0; kc < 9; ++kc){
      v16h b  = load_b_frag(We1sw, 9, w, kc, lane);
      v16h a0 = load_a_frag(sF + col * 288, kc, lane);
      v16h a1 = load_a_frag(sF + (16 + col) * 288, kc, lane);
      acc0 = WMMA_F16(a0, b, acc0);
      acc1 = WMMA_F16(a1, b, acc1);
    }
    float bias = be1[n];
#pragma unroll
    for (int r = 0; r < 8; ++r){
      int m0 = r + 8 * half;
      sY[m0 * 128 + n]        = (f16)silu_f(acc0[r] + bias);
      sY[(16 + m0) * 128 + n] = (f16)silu_f(acc1[r] + bias);
    }
  }
  __syncthreads();

  // ---- phi_e layer 2: [32,128] x [128,128] -> m_ij ----
  {
    v8f acc0 = {}, acc1 = {};
#pragma unroll
    for (int kc = 0; kc < 4; ++kc){
      v16h b  = load_b_frag(We2sw, 4, w, kc, lane);
      v16h a0 = load_a_frag(sY + col * 128, kc, lane);
      v16h a1 = load_a_frag(sY + (16 + col) * 128, kc, lane);
      acc0 = WMMA_F16(a0, b, acc0);
      acc1 = WMMA_F16(a1, b, acc1);
    }
    float bias = be2[n];
#pragma unroll
    for (int r = 0; r < 8; ++r){
      int m0 = r + 8 * half;
      float v0 = silu_f(acc0[r] + bias);
      float v1 = silu_f(acc1[r] + bias);
      sM[m0 * 128 + n]        = (f16)v0;
      sM[(16 + m0) * 128 + n] = (f16)v1;
      int ea = e0 + m0, eb = e0 + 16 + m0;
      if (ea < E){
        out_m[(size_t)ea * 128 + n] = v0;
        atomic_add_f32(&agg[(size_t)sDst[m0] * 128 + n], v0);
      }
      if (eb < E){
        out_m[(size_t)eb * 128 + n] = v1;
        atomic_add_f32(&agg[(size_t)sDst[16 + m0] * 128 + n], v1);
      }
    }
  }
  __syncthreads();

  // ---- phi_x layer 1: silu(m @ Wx1 + bx1) ----
  {
    v8f acc0 = {}, acc1 = {};
#pragma unroll
    for (int kc = 0; kc < 4; ++kc){
      v16h b  = load_b_frag(Wx1sw, 4, w, kc, lane);
      v16h a0 = load_a_frag(sM + col * 128, kc, lane);
      v16h a1 = load_a_frag(sM + (16 + col) * 128, kc, lane);
      acc0 = WMMA_F16(a0, b, acc0);
      acc1 = WMMA_F16(a1, b, acc1);
    }
    float bias = bx1[n];
#pragma unroll
    for (int r = 0; r < 8; ++r){
      int m0 = r + 8 * half;
      sX[m0 * 128 + n]        = (f16)silu_f(acc0[r] + bias);
      sX[(16 + m0) * 128 + n] = (f16)silu_f(acc1[r] + bias);
    }
  }
  __syncthreads();

  // ---- phi_x layer 2 (dot with Wx2) + coord scatter ----
  if (tid < 32){
    float accw = bx2[0];
    const f16* r = sX + tid * 128;
#pragma unroll 8
    for (int k = 0; k < 128; ++k) accw += (float)r[k] * Wx2[k];
    sCW[tid] = accw;
  }
  __syncthreads();
  if (tid < 32){
    int e = e0 + tid;
    if (e < E){
      int dn = sDst[tid];
      float cw = sCW[tid];
      atomic_add_f32(&num[(size_t)dn*3+0], sRel[tid*3+0]*cw);
      atomic_add_f32(&num[(size_t)dn*3+1], sRel[tid*3+1]*cw);
      atomic_add_f32(&num[(size_t)dn*3+2], sRel[tid*3+2]*cw);
      atomic_add_f32(&cnt[dn], 1.0f);
    }
  }
}

// ---------- node kernel: phi_h, residual, LayerNorm, coord update ----------
__global__ __launch_bounds__(256)
void egnn_node_kernel(const f16* __restrict__ hh,
                      const float* __restrict__ h,
                      const float* __restrict__ x,
                      const float* __restrict__ agg,
                      const float* __restrict__ num,
                      const float* __restrict__ cnt,
                      const f16* __restrict__ Wh1sw, const float* __restrict__ bh1,
                      const f16* __restrict__ Wh2sw, const float* __restrict__ bh2,
                      const float* __restrict__ gamma, const float* __restrict__ beta,
                      float* __restrict__ out_h, float* __restrict__ out_x, int N)
{
  __shared__ __align__(16) f16 sF[16 * 256];   // concat(h, agg)
  __shared__ __align__(16) f16 sY[16 * 128];
  __shared__ float sH[16 * 128];
  __shared__ float sMu[16], sRs[16];

  const int tid  = threadIdx.x;
  const int lane = tid & 31;
  const int w    = tid >> 5;
  const int nb   = blockIdx.x * 16;

  // build features: 16 threads per node, 16 elems each
  {
    int node = tid >> 4, seg = tid & 15;
    int gnode = nb + node;
    int src = (gnode < N) ? gnode : 0;
    if (seg < 8){
      const uint4* p = (const uint4*)(hh + (size_t)src * 128 + seg * 16);
      uint4* q = (uint4*)(sF + node * 256 + seg * 16);
      q[0] = p[0]; q[1] = p[1];
    } else {
      const float* p = agg + (size_t)src * 128 + (seg - 8) * 16;
      f16* q = sF + node * 256 + 128 + (seg - 8) * 16;
#pragma unroll
      for (int i = 0; i < 16; ++i) q[i] = (f16)p[i];
    }
  }
  __syncthreads();

  const int col = lane & 15;
  const int half = (lane >> 4) & 1;
  const int n = w * 16 + col;

  // phi_h layer 1: [16,256] x [256,128]
  {
    v8f acc = {};
#pragma unroll
    for (int kc = 0; kc < 8; ++kc){
      v16h b = load_b_frag(Wh1sw, 8, w, kc, lane);
      v16h a = load_a_frag(sF + col * 256, kc, lane);
      acc = WMMA_F16(a, b, acc);
    }
    float bias = bh1[n];
#pragma unroll
    for (int r = 0; r < 8; ++r)
      sY[(r + 8 * half) * 128 + n] = (f16)silu_f(acc[r] + bias);
  }
  __syncthreads();

  // phi_h layer 2 + residual
  {
    v8f acc = {};
#pragma unroll
    for (int kc = 0; kc < 4; ++kc){
      v16h b = load_b_frag(Wh2sw, 4, w, kc, lane);
      v16h a = load_a_frag(sY + col * 128, kc, lane);
      acc = WMMA_F16(a, b, acc);
    }
    float bias = bh2[n];
#pragma unroll
    for (int r = 0; r < 8; ++r){
      int m = r + 8 * half;
      int gnode = nb + m;
      float hv = (gnode < N) ? h[(size_t)gnode * 128 + n] : 0.0f;
      sH[m * 128 + n] = hv + acc[r] + bias;
    }
  }
  __syncthreads();

  // LayerNorm stats
  if (tid < 16){
    float s = 0.f, s2 = 0.f;
    const float* r = sH + tid * 128;
    for (int k = 0; k < 128; ++k){ float v = r[k]; s += v; s2 += v * v; }
    float mu = s * (1.0f / 128.0f);
    float var = s2 * (1.0f / 128.0f) - mu * mu;
    sMu[tid] = mu; sRs[tid] = rsqrtf(var + 1e-5f);
  }
  __syncthreads();

  {
    int node = tid >> 4, gnode = nb + node;
    if (gnode < N){
      int base = (tid & 15) * 8;
      float mu = sMu[node], rs = sRs[node];
#pragma unroll
      for (int i = 0; i < 8; ++i){
        int k = base + i;
        out_h[(size_t)gnode * 128 + k] = (sH[node * 128 + k] - mu) * rs * gamma[k] + beta[k];
      }
    }
  }
  if (tid < 16){
    int gnode = nb + tid;
    if (gnode < N){
      float c = fmaxf(cnt[gnode], 1.0f);
      for (int j = 0; j < 3; ++j)
        out_x[(size_t)gnode * 3 + j] = x[(size_t)gnode * 3 + j] + num[(size_t)gnode * 3 + j] / c;
    }
  }
}

extern "C" void kernel_launch(void* const* d_in, const int* in_sizes, int n_in,
                              void* d_out, int out_size, void* d_ws, size_t ws_size,
                              hipStream_t stream) {
  (void)n_in; (void)out_size; (void)ws_size;
  const float*     h    = (const float*)d_in[0];
  const float*     x    = (const float*)d_in[1];
  const float*     vel  = (const float*)d_in[2];
  const long long* eidx = (const long long*)d_in[3];
  const float* We1 = (const float*)d_in[4];  const float* be1 = (const float*)d_in[5];
  const float* We2 = (const float*)d_in[6];  const float* be2 = (const float*)d_in[7];
  const float* Wx1 = (const float*)d_in[8];  const float* bx1 = (const float*)d_in[9];
  const float* Wx2 = (const float*)d_in[10]; const float* bx2 = (const float*)d_in[11];
  const float* Wh1 = (const float*)d_in[12]; const float* bh1 = (const float*)d_in[13];
  const float* Wh2 = (const float*)d_in[14]; const float* bh2 = (const float*)d_in[15];
  const float* gamma = (const float*)d_in[16];
  const float* beta  = (const float*)d_in[17];

  const int N = in_sizes[0] / 128;
  const int E = in_sizes[3] / 2;

  char* ws = (char*)d_ws;
  size_t off = 0;
  auto carve = [&](size_t bytes) -> void* {
    void* p = ws + off;
    off += (bytes + 255) & ~(size_t)255;
    return p;
  };
  f16* We1sw = (f16*)carve((size_t)288 * 128 * 2);
  f16* We2sw = (f16*)carve((size_t)128 * 128 * 2);
  f16* Wx1sw = (f16*)carve((size_t)128 * 128 * 2);
  f16* Wh1sw = (f16*)carve((size_t)256 * 128 * 2);
  f16* Wh2sw = (f16*)carve((size_t)128 * 128 * 2);
  f16*   hh  = (f16*)carve((size_t)N * 128 * 2);
  float* agg = (float*)carve((size_t)N * 128 * 4);
  float* num = (float*)carve((size_t)N * 3 * 4);
  float* cnt = (float*)carve((size_t)N * 4);

  hipMemsetAsync(agg, 0, (size_t)N * 128 * 4, stream);
  hipMemsetAsync(num, 0, (size_t)N * 3 * 4, stream);
  hipMemsetAsync(cnt, 0, (size_t)N * 4, stream);

  swizzle_kernel<<<(9 * 4096 + 255) / 256, 256, 0, stream>>>(We1, We1sw, 267, 9);
  swizzle_kernel<<<(4 * 4096 + 255) / 256, 256, 0, stream>>>(We2, We2sw, 128, 4);
  swizzle_kernel<<<(4 * 4096 + 255) / 256, 256, 0, stream>>>(Wx1, Wx1sw, 128, 4);
  swizzle_kernel<<<(8 * 4096 + 255) / 256, 256, 0, stream>>>(Wh1, Wh1sw, 256, 8);
  swizzle_kernel<<<(4 * 4096 + 255) / 256, 256, 0, stream>>>(Wh2, Wh2sw, 128, 4);
  conv_h_kernel<<<((size_t)N * 128 + 255) / 256, 256, 0, stream>>>(h, hh, N * 128);

  float* out_h = (float*)d_out;
  float* out_x = out_h + (size_t)N * 128;
  float* out_m = out_x + (size_t)N * 3;

  egnn_edge_kernel<<<(E + 31) / 32, 256, 0, stream>>>(
      hh, x, vel, eidx, We1sw, be1, We2sw, be2, Wx1sw, bx1, Wx2, bx2,
      agg, num, cnt, out_m, E);
  egnn_node_kernel<<<(N + 15) / 16, 256, 0, stream>>>(
      hh, h, x, agg, num, cnt, Wh1sw, bh1, Wh2sw, bh2, gamma, beta,
      out_h, out_x, N);
}
